// EuclideanCodebook_6150393168577
// MI455X (gfx1250) — matrix operand order repaired and checked by reading, not compiled, not run on hardware
//
#include <hip/hip_runtime.h>

typedef __attribute__((ext_vector_type(16))) __bf16 v16bf;
typedef __attribute__((ext_vector_type(8)))  float  v8f;

using u16 = unsigned short;
using u32 = unsigned int;
using u64 = unsigned long long;

typedef __attribute__((address_space(3))) char lds_char;

union BF16x16 { uint4 u[2]; v16bf v; };

#define N_ROWS 65536
#define DIM    512
#define KCODES 2048
#define CB     64              // codes per LDS buffer (2 buffers, ping-pong)
#define NCB    (KCODES / CB)   // 32 blocks
#define KSTEPS (DIM / 32)      // 16 wmma k-steps over D

__device__ __forceinline__ void sched_split() {
#if __has_builtin(__builtin_amdgcn_sched_barrier)
  __builtin_amdgcn_sched_barrier(0);   // nothing crosses: loads stay hoisted
#else
  asm volatile("" ::: "memory");
#endif
}

__device__ __forceinline__ u16 f2bf(float f) {
  u32 u = __float_as_uint(f);
  u32 r = u + 0x7fffu + ((u >> 16) & 1u);   // round-to-nearest-even
  return (u16)(r >> 16);
}

// ---------------- convert x: f32 -> bf16 (one float4 per thread) -------------
__global__ void cvt_x_kernel(const float* __restrict__ x, u16* __restrict__ xb) {
  int t = blockIdx.x * blockDim.x + threadIdx.x;
  float4 v = ((const float4*)x)[t];
  u32 lo = (u32)f2bf(v.x) | ((u32)f2bf(v.y) << 16);
  u32 hi = (u32)f2bf(v.z) | ((u32)f2bf(v.w) << 16);
  ((uint2*)xb)[t] = make_uint2(lo, hi);
}

// ------------- convert embed: f32 -> bf16, and e_sq[k] = |e_k|^2 -------------
__global__ void cvt_embed_kernel(const float* __restrict__ e, u16* __restrict__ eb,
                                 float* __restrict__ esq) {
  __shared__ float part[4];
  const int row = blockIdx.x;          // one codebook row per block (128 thr)
  const int t   = threadIdx.x;
  float4 v = ((const float4*)(e + (size_t)row * DIM))[t];
  u32 lo = (u32)f2bf(v.x) | ((u32)f2bf(v.y) << 16);
  u32 hi = (u32)f2bf(v.z) | ((u32)f2bf(v.w) << 16);
  ((uint2*)(eb + (size_t)row * DIM))[t] = make_uint2(lo, hi);
  float s = v.x * v.x + v.y * v.y + v.z * v.z + v.w * v.w;
  #pragma unroll
  for (int m = 16; m >= 1; m >>= 1) s += __shfl_xor(s, m, 32);
  if ((t & 31) == 0) part[t >> 5] = s;
  __syncthreads();
  if (t == 0) esq[row] = part[0] + part[1] + part[2] + part[3];
}

// --------------------- fused GEMM + argmax + gather --------------------------
// 256 threads = 8 waves; each wave owns a 16-row stripe of x; WG covers 128 rows.
// LDS ping-pong: async-DMA block n+1 into one 64KB buffer while WMMAs consume
// the other; s_wait_asynccnt 0 + barrier only at the end of each block.
__global__ __launch_bounds__(256)
void vq_argmax_kernel(const u16* __restrict__ xb, const u16* __restrict__ eb,
                      const float* __restrict__ esq, const float* __restrict__ embed,
                      float* __restrict__ out) {
  extern __shared__ char smem[];
  u16* sEmb = (u16*)smem;                                   // 2 x CB*DIM bf16
  int* sIdx = (int*)(smem + 2 * (size_t)CB * DIM * 2);      // 8*16 ints

  const int tid  = threadIdx.x;
  const int lane = tid & 31;
  const int wave = tid >> 5;
  const int col  = lane & 15;   // C-matrix column slot for this lane
  const int half = lane >> 4;   // 0: rows 0..7, 1: rows 8..15 of the stripe
  const int wgRow = blockIdx.x * 128;
  const int aRow  = wgRow + wave * 16 + col;   // A row this lane supplies

  // Preload the full A stripe (16 rows x 512 bf16) into 128 VGPRs, ISA layout:
  // lane<16 holds K {0..7,16..23}+32*kk, lane>=16 holds K {8..15,24..31}+32*kk.
  BF16x16 a[KSTEPS];
  {
    const u16* xr = xb + (size_t)aRow * DIM + half * 8;
    #pragma unroll
    for (int kk = 0; kk < KSTEPS; ++kk) {
      a[kk].u[0] = *(const uint4*)(xr + kk * 32);
      a[kk].u[1] = *(const uint4*)(xr + kk * 32 + 16);
    }
  }

  float best[8];
  int   bidx[8];
  #pragma unroll
  for (int i = 0; i < 8; ++i) { best[i] = -3.4e38f; bidx[i] = 0; }

  const u32 ldsEmb = (u32)(u64)(lds_char*)sEmb;  // LDS byte offset of sEmb

  // issue one block's worth of async memory->LDS DMA (no VGPR round trip)
  auto stage = [&](int cb, int buf) {
    const u64 gbase = (u64)(const void*)(eb + (size_t)cb * CB * DIM);
    const u32 lbase = ldsEmb + (u32)buf * (CB * DIM * 2);
    #pragma unroll
    for (int t = 0; t < (CB * DIM * 2 / 16) / 256; ++t) {   // 16 b128s/thread
      u32 off = (u32)(tid + t * 256) * 16u;
      asm volatile("global_load_async_to_lds_b128 %0, %1, %2"
                   :: "v"(lbase + off), "v"(off), "s"(gbase) : "memory");
    }
  };

  stage(0, 0);                                         // prologue
  asm volatile("s_wait_asynccnt 0" ::: "memory");
  __syncthreads();

  for (int cb = 0; cb < NCB; ++cb) {
    if (cb + 1 < NCB) stage(cb + 1, (cb + 1) & 1);     // overlap DMA w/ WMMA
    if (cb + 2 < NCB)                                  // warm L2 for block n+2
      __builtin_prefetch((const char*)(eb + (size_t)(cb + 2) * CB * DIM) + tid * 256, 0, 1);

    const u16* sbuf = sEmb + (size_t)(cb & 1) * CB * DIM;
    for (int ct = 0; ct < CB / 16; ++ct) {             // 4 code tiles of 16
      const u16* bbase = sbuf + (size_t)(ct * 16 + col) * DIM + half * 8;

      // Phase 1: bulk-load ALL 16 B fragments (32 ds_load_b128, 128 VGPRs).
      // Every b[kk] stays live until its WMMA, so RA cannot coalesce the
      // windows; the counter pass emits staggered s_wait_dscnt as the chain
      // consumes fragments, keeping up to 30 LDS ops in flight under XDL.
      BF16x16 b[KSTEPS];
      #pragma unroll
      for (int kk = 0; kk < KSTEPS; ++kk) {
        b[kk].u[0] = *(const uint4*)(bbase + kk * 32);
        b[kk].u[1] = *(const uint4*)(bbase + kk * 32 + 16);
      }
      sched_split();   // scheduler may not sink loads into the WMMA chain

      // Phase 2: 16-step WMMA accumulation chain.
      v8f acc = {};
      #pragma unroll
      for (int kk = 0; kk < KSTEPS; ++kk)
        acc = __builtin_amdgcn_wmma_f32_16x16x32_bf16(
            false, a[kk].v, false, b[kk].v, (short)0, acc, false, false);

      const int   code = cb * CB + ct * 16 + col;
      const float eq   = esq[code];
      #pragma unroll
      for (int i = 0; i < 8; ++i) {        // score = 2*x.e - |e|^2 per row
        float s = 2.0f * acc[i] - eq;
        if (s > best[i]) { best[i] = s; bidx[i] = code; }
      }
    }

    // seal the block: next-block DMA done for everyone, this buffer consumed
    asm volatile("s_wait_asynccnt 0" ::: "memory");
    __syncthreads();
  }

  // cross-lane argmax across the 16 column-slots of each half (rows disjoint)
  #pragma unroll
  for (int i = 0; i < 8; ++i) {
    float v = best[i]; int ix = bidx[i];
    #pragma unroll
    for (int m = 8; m >= 1; m >>= 1) {
      float ov = __shfl_xor(v, m, 32);
      int   oi = __shfl_xor(ix, m, 32);
      if (ov > v || (ov == v && oi < ix)) { v = ov; ix = oi; }
    }
    best[i] = v; bidx[i] = ix;
  }

  if (col == 0) {           // lane 0 -> rows 0..7, lane 16 -> rows 8..15
    #pragma unroll
    for (int i = 0; i < 8; ++i) {
      int r   = half * 8 + i;
      int row = wgRow + wave * 16 + r;
      out[row] = (float)bidx[i];
      sIdx[wave * 16 + r] = bidx[i];
    }
  }
  asm volatile("s_wait_dscnt 0" ::: "memory");   // intra-wave LDS handoff

  // gather: decode 16 rows/wave from the original f32 codebook (exact)
  float* outq = out + N_ROWS;
  for (int r = 0; r < 16; ++r) {
    int idx = sIdx[wave * 16 + r];
    const float4* src = (const float4*)(embed + (size_t)idx * DIM);
    float4* dst = (float4*)(outq + (size_t)(wgRow + wave * 16 + r) * DIM);
    #pragma unroll
    for (int j = 0; j < 4; ++j)
      dst[lane + j * 32] = src[lane + j * 32];
  }
}

extern "C" void kernel_launch(void* const* d_in, const int* in_sizes, int n_in,
                              void* d_out, int out_size, void* d_ws, size_t ws_size,
                              hipStream_t stream) {
  const float* x     = (const float*)d_in[0];
  const float* embed = (const float*)d_in[1];
  float* out = (float*)d_out;

  u16*   xb  = (u16*)d_ws;                                                 // 64 MB
  u16*   eb  = (u16*)((char*)d_ws + (size_t)N_ROWS * DIM * 2);             // 2 MB
  float* esq = (float*)((char*)d_ws + (size_t)N_ROWS * DIM * 2
                                     + (size_t)KCODES * DIM * 2);          // 8 KB

  cvt_x_kernel<<<(N_ROWS * DIM / 4) / 256, 256, 0, stream>>>(x, xb);
  cvt_embed_kernel<<<KCODES, 128, 0, stream>>>(embed, eb, esq);

  const size_t smem = 2 * (size_t)CB * DIM * 2 + 8 * 16 * 4;  // 128 KB + idx
  (void)hipFuncSetAttribute(reinterpret_cast<const void*>(vq_argmax_kernel),
                            hipFuncAttributeMaxDynamicSharedMemorySize, (int)smem);
  vq_argmax_kernel<<<N_ROWS / 128, 256, smem, stream>>>(xb, eb, esq, embed, out);
}
